// GRU_55937654063623
// MI455X (gfx1250) — matrix operand order repaired and checked by reading, not compiled, not run on hardware
//
#include <hip/hip_runtime.h>
#include <hip/hip_bf16.h>

#define BB 32
#define TT 1024
#define DD 512
#define HH 1024
#define BT (BB*TT)
#define EPSV 1e-5f

typedef __bf16 bf16;
typedef __attribute__((ext_vector_type(16))) __bf16 v16bf;
typedef __attribute__((ext_vector_type(8)))  __bf16 v8bf;
typedef __attribute__((ext_vector_type(8)))  float  v8f;

// ---------------- workspace layout (bytes) ----------------
constexpr size_t SZ_XG   = (size_t)BT * HH * 4;                 // 128 MB each
constexpr size_t OFF_XZ  = 0;
constexpr size_t OFF_XR  = OFF_XZ + SZ_XG;
constexpr size_t OFF_XH  = OFF_XR + SZ_XG;
constexpr size_t OFF_XBF = OFF_XH + SZ_XG;                      // x in bf16
constexpr size_t OFF_WP  = OFF_XBF + (size_t)BT*DD*2;           // packed W (3)
constexpr size_t OFF_UP  = OFF_WP  + (size_t)3*DD*HH*2;         // packed U (3)
constexpr size_t OFF_HF  = OFF_UP  + (size_t)3*HH*HH*2;         // h fp32
constexpr size_t OFF_HBF = OFF_HF  + (size_t)BB*HH*4;           // h bf16
constexpr size_t OFF_ACC = OFF_HBF + (size_t)BB*HH*2;           // h@U per gate
constexpr size_t OFF_ZB  = OFF_ACC + (size_t)3*BB*HH*4;         // z buffer
constexpr size_t OFF_AHB = OFF_ZB  + (size_t)BB*HH*4;           // ah buffer
constexpr size_t OFF_PS  = OFF_AHB + (size_t)BB*HH*4;           // partial sums z,r
constexpr size_t OFF_PQ  = OFF_PS  + (size_t)2*BB*64*4;
constexpr size_t OFF_P2S = OFF_PQ  + (size_t)2*BB*64*4;         // partials for ah
constexpr size_t OFF_P2Q = OFF_P2S + (size_t)BB*4*4;
constexpr size_t OFF_ST  = OFF_P2Q + (size_t)BB*4*4;            // (mean,rstd) x 3 gates
constexpr size_t OFF_BAR = OFF_ST  + (size_t)3*BB*2*4;          // barrier {cnt,gen}

// ---------------- async LDS staging (CDNA5 GLOBAL_LOAD_ASYNC_TO_LDS) ----------------
#if defined(__has_builtin)
#if __has_builtin(__builtin_amdgcn_global_load_async_to_lds_b128)
#define HAVE_ASYNC_LDS 1
#endif
#endif

#ifdef HAVE_ASYNC_LDS
// builtin expects: (int4 __device__ *  i.e. AS(1), int4 __shared__ * i.e. AS(3), imm off, imm cpol)
typedef int v4i __attribute__((__vector_size__(4 * sizeof(int))));
typedef __attribute__((address_space(1))) v4i* as1_v4i;
typedef __attribute__((address_space(3))) v4i* as3_v4i;
__device__ inline void async_cp16(const bf16* gp, bf16* lp) {
  // global -> LDS, 16 bytes per active lane, tracked by ASYNCcnt
  __builtin_amdgcn_global_load_async_to_lds_b128(
      (as1_v4i)(uintptr_t)gp, (as3_v4i)(unsigned)(uintptr_t)lp, 0, 0);
}
__device__ inline void async_wait0() {
#if __has_builtin(__builtin_amdgcn_s_wait_asynccnt)
  __builtin_amdgcn_s_wait_asynccnt(0);
#else
  asm volatile("s_wait_asynccnt 0" ::: "memory");
#endif
}
#else
__device__ inline void async_cp16(const bf16* gp, bf16* lp) {
  *(v8bf*)lp = *(const v8bf*)gp;          // fallback: vector load + ds_store_b128
}
__device__ inline void async_wait0() {}
#endif

// ---------------- helpers ----------------
__device__ inline bf16 f2bf(float f) {
  unsigned u = __float_as_uint(f);
  unsigned r = (u + 0x7fffu + ((u >> 16) & 1u)) >> 16;
  unsigned short s = (unsigned short)r;
  return *(bf16*)&s;
}
__device__ inline float sigmoidf(float x) { return 1.0f / (1.0f + __expf(-x)); }

// A-fragment (16x32 bf16) from a row-major matrix, per ISA 7.12.2:
// lanes 0-15 row M=lane : v0..3 = K 0..7,  v4..7 = K 16..23
// lanes16-31 row M=lane-16: v0..3 = K 8..15, v4..7 = K 24..31
__device__ inline v16bf load_a_frag(const bf16* base, int lda, int m0, int k0, int lane) {
  int row  = m0 + (lane & 15);
  int half = lane >> 4;
  const bf16* p = base + (size_t)row * lda + k0 + half * 8;
  v8bf lo = *(const v8bf*)(p);
  v8bf hi = *(const v8bf*)(p + 16);
  v16bf a;
#pragma unroll
  for (int i = 0; i < 8; ++i) { a[i] = lo[i]; a[i + 8] = hi[i]; }
  return a;
}

// software grid barrier (persistent kernel)
__device__ inline void gbar(int* cnt, int* gen, int nwg) {
  __syncthreads();
  if (threadIdx.x == 0) {
    __threadfence();
    int g = __hip_atomic_load(gen, __ATOMIC_ACQUIRE, __HIP_MEMORY_SCOPE_AGENT);
    if (atomicAdd(cnt, 1) == nwg - 1) {
      atomicExch(cnt, 0);
      __hip_atomic_store(gen, g + 1, __ATOMIC_RELEASE, __HIP_MEMORY_SCOPE_AGENT);
    } else {
      while (__hip_atomic_load(gen, __ATOMIC_ACQUIRE, __HIP_MEMORY_SCOPE_AGENT) == g)
        __builtin_amdgcn_s_sleep(2);
    }
    __threadfence();
  }
  __syncthreads();
}

// ---------------- prep kernels ----------------
__global__ void k_cvt_x(const float* __restrict__ x, bf16* __restrict__ xbf, int n) {
  int i = blockIdx.x * blockDim.x + threadIdx.x;
  if (i < n) xbf[i] = f2bf(x[i]);
}

// pack W into B-fragment layout: [(g*64+nt)*16+kt]*512 + lane*16 + e
// element e of lane -> K = kt*32 + 8*(lane>>4) + (e&7) + 16*(e>>3), N = nt*16 + (lane&15)
__global__ void k_pack_w(const float* __restrict__ Wz, const float* __restrict__ Wr,
                         const float* __restrict__ Wh, bf16* __restrict__ wp) {
  int idx = blockIdx.x * blockDim.x + threadIdx.x;
  if (idx >= 3 * 64 * 16 * 512) return;
  int e    = idx & 15;
  int lane = (idx >> 4) & 31;
  int kt   = (idx >> 9) & 15;
  int nt   = (idx >> 13) & 63;
  int g    = idx >> 19;
  int K = kt * 32 + (lane >> 4) * 8 + (e & 7) + (e >> 3) * 16;
  int N = nt * 16 + (lane & 15);
  const float* W = (g == 0) ? Wz : ((g == 1) ? Wr : Wh);
  wp[idx] = f2bf(W[(size_t)K * HH + N]);
}

__global__ void k_pack_u(const float* __restrict__ Uz, const float* __restrict__ Ur,
                         const float* __restrict__ Uh, bf16* __restrict__ up) {
  int idx = blockIdx.x * blockDim.x + threadIdx.x;
  if (idx >= 3 * 64 * 32 * 512) return;
  int e    = idx & 15;
  int lane = (idx >> 4) & 31;
  int kt   = (idx >> 9) & 31;
  int nt   = (idx >> 14) & 63;
  int g    = idx >> 20;
  int K = kt * 32 + (lane >> 4) * 8 + (e & 7) + (e >> 3) * 16;
  int N = nt * 16 + (lane & 15);
  const float* U = (g == 0) ? Uz : ((g == 1) ? Ur : Uh);
  up[idx] = f2bf(U[(size_t)K * HH + N]);
}

__global__ void k_init(const float* __restrict__ h0, float* __restrict__ hf,
                       bf16* __restrict__ hbf, int* __restrict__ bar) {
  int i = blockIdx.x * blockDim.x + threadIdx.x;
  if (i < BB * HH) { float v = h0[i]; hf[i] = v; hbf[i] = f2bf(v); }
  if (i < 2) bar[i] = 0;
}

// ---------------- input projection: xg = x @ W_g + bias ----------------
// One block = 8 waves sharing one 32-row x strip (staged once into LDS).
__global__ void __launch_bounds__(256)
k_proj(const bf16* __restrict__ xbf, const bf16* __restrict__ wp,
       const float* __restrict__ b_z, const float* __restrict__ b_r,
       float* __restrict__ xz, float* __restrict__ xr, float* __restrict__ xh) {
  __shared__ bf16 xs[32 * DD];                       // 32 KB

  const int base = blockIdx.x * 8;                   // 8 wave-tasks per block
  const int g    = base >> 16;                       // uniform within block
  const int rem  = base & 65535;
  const int mt   = rem >> 6;                         // uniform within block
  const int ntb  = rem & 63;
  const int w    = threadIdx.x >> 5;
  const int lane = threadIdx.x & 31;
  const int nt   = ntb + w;

  // stage A strip (32 x 512 bf16) into LDS: 2048 x 16B chunks / 256 threads
  {
    const bf16* gsrc = xbf + (size_t)mt * 32 * DD;
#pragma unroll
    for (int i = 0; i < 8; ++i) {
      int c = threadIdx.x + i * 256;
      async_cp16(gsrc + c * 8, xs + c * 8);
    }
    async_wait0();
  }
  __syncthreads();

  const bf16* wpb = wp + (size_t)(g * 64 + nt) * 16 * 512;
  v8f c0 = {}, c1 = {};
#pragma unroll 4
  for (int kt = 0; kt < 16; ++kt) {
    __builtin_prefetch(wpb + (kt + 1) * 512 + lane * 16, 0, 0);
    v16bf bfr = *(const v16bf*)(wpb + kt * 512 + lane * 16);
    v16bf a0  = load_a_frag(xs, DD, 0,  kt * 32, lane);
    v16bf a1  = load_a_frag(xs, DD, 16, kt * 32, lane);
    c0 = __builtin_amdgcn_wmma_f32_16x16x32_bf16(false, a0, false, bfr, (short)0, c0, false, false);
    c1 = __builtin_amdgcn_wmma_f32_16x16x32_bf16(false, a1, false, bfr, (short)0, c1, false, false);
  }
  int col  = nt * 16 + (lane & 15);
  int half = lane >> 4;
  float bv = ((g == 1) ? b_r : b_z)[col];   // gate h reuses b_z (as in reference)
  float* o = (g == 0) ? xz : ((g == 1) ? xr : xh);
#pragma unroll
  for (int v = 0; v < 8; ++v) {
    o[((size_t)mt * 32 + half * 8 + v) * HH + col]      = c0[v] + bv;
    o[((size_t)mt * 32 + 16 + half * 8 + v) * HH + col] = c1[v] + bv;
  }
}

// ---------------- persistent recurrent scan ----------------
// grid: 48 blocks x 128 threads = 192 waves = 3 gates x 64 n-tiles
#define NWG 48
__global__ void __launch_bounds__(128)
k_scan(bf16* __restrict__ hbf, float* __restrict__ hf,
       const bf16* __restrict__ up,
       const float* __restrict__ xz, const float* __restrict__ xr, const float* __restrict__ xh,
       float* __restrict__ acc, float* __restrict__ zb, float* __restrict__ ahb,
       float* __restrict__ ps, float* __restrict__ pq,
       float* __restrict__ p2s, float* __restrict__ p2q,
       float* __restrict__ st, int* __restrict__ bar,
       const float* __restrict__ g_z, const float* __restrict__ be_z,
       const float* __restrict__ g_r, const float* __restrict__ be_r,
       const float* __restrict__ g_h, const float* __restrict__ be_h,
       float* __restrict__ out, float* __restrict__ hlast) {
  __shared__ bf16 hsm[BB * HH];                      // 64 KB, whole h per block

  const int wid  = blockIdx.x * 4 + (threadIdx.x >> 5);
  const int lane = threadIdx.x & 31;
  const int g    = wid >> 6;                         // 0..2 gate
  const int nt   = wid & 63;                         // n-tile
  int* cnt = bar;
  int* gen = bar + 1;

  for (int t = 0; t < TT; ++t) {
    // ---- stage h (32x1024 bf16) into LDS: 4096 x 16B chunks / 128 threads ----
#pragma unroll
    for (int i = 0; i < 32; ++i) {
      int c = threadIdx.x + i * 128;
      async_cp16(hbf + c * 8, hsm + c * 8);
    }
    async_wait0();
    __syncthreads();

    // ---- phase G: acc[g] = h @ U_g, plus row-partials for gates z,r ----
    {
      const bf16* upb = up + (size_t)(g * 64 + nt) * 32 * 512;
      v8f c0 = {}, c1 = {};
#pragma unroll 4
      for (int kt = 0; kt < 32; ++kt) {
        __builtin_prefetch(upb + (kt + 1) * 512 + lane * 16, 0, 0);
        v16bf bfr = *(const v16bf*)(upb + kt * 512 + lane * 16);
        v16bf a0  = load_a_frag(hsm, HH, 0,  kt * 32, lane);
        v16bf a1  = load_a_frag(hsm, HH, 16, kt * 32, lane);
        c0 = __builtin_amdgcn_wmma_f32_16x16x32_bf16(false, a0, false, bfr, (short)0, c0, false, false);
        c1 = __builtin_amdgcn_wmma_f32_16x16x32_bf16(false, a1, false, bfr, (short)0, c1, false, false);
      }
      const int col  = nt * 16 + (lane & 15);
      const int half = lane >> 4;
      float* accg = acc + (size_t)g * BB * HH;
#pragma unroll
      for (int v = 0; v < 8; ++v) {
        accg[(half * 8 + v) * HH + col]      = c0[v];
        accg[(16 + half * 8 + v) * HH + col] = c1[v];
      }
      if (g < 2) {  // row partial sums over this wave's 16 columns (z, r gates)
        float s0[8], q0[8], s1[8], q1[8];
#pragma unroll
        for (int v = 0; v < 8; ++v) {
          s0[v] = c0[v]; q0[v] = c0[v] * c0[v];
          s1[v] = c1[v]; q1[v] = c1[v] * c1[v];
        }
#pragma unroll
        for (int m = 1; m < 16; m <<= 1) {
#pragma unroll
          for (int v = 0; v < 8; ++v) {
            s0[v] += __shfl_xor(s0[v], m, 32); q0[v] += __shfl_xor(q0[v], m, 32);
            s1[v] += __shfl_xor(s1[v], m, 32); q1[v] += __shfl_xor(q1[v], m, 32);
          }
        }
        if ((lane & 15) == 0) {
          int mb = half * 8;
#pragma unroll
          for (int v = 0; v < 8; ++v) {
            ps[((size_t)g * BB + mb + v) * 64 + nt]      = s0[v];
            pq[((size_t)g * BB + mb + v) * 64 + nt]      = q0[v];
            ps[((size_t)g * BB + 16 + mb + v) * 64 + nt] = s1[v];
            pq[((size_t)g * BB + 16 + mb + v) * 64 + nt] = q1[v];
          }
        }
      }
    }
    gbar(cnt, gen, NWG);

    // ---- phase S1: LN stats for z,r rows over (xg + hU) ----
    // hU moments come from GEMM partials; x moments and cross terms accumulated here:
    // sum((x+u)^2) = sum u^2 + sum x^2 + 2*sum(x*u)
    if (wid < 64) {
      const int gg = wid >> 5, b = wid & 31;
      const float* psr = ps + (size_t)(gg * BB + b) * 64;
      const float* pqr = pq + (size_t)(gg * BB + b) * 64;
      float s = psr[lane] + psr[lane + 32];
      float q = pqr[lane] + pqr[lane + 32];
      const float* xg = ((gg == 0) ? xz : xr) + ((size_t)b * TT + t) * HH;
      const float* ar = acc + ((size_t)gg * BB + b) * HH;
      float s2 = 0.f, q2 = 0.f;
#pragma unroll
      for (int i = 0; i < 32; ++i) {
        int c = i * 32 + lane;
        float xv = xg[c];
        float uv = ar[c];
        s2 += xv;
        q2 += xv * xv + 2.f * xv * uv;
      }
#pragma unroll
      for (int m = 1; m < 32; m <<= 1) {
        s  += __shfl_xor(s,  m, 32); q  += __shfl_xor(q,  m, 32);
        s2 += __shfl_xor(s2, m, 32); q2 += __shfl_xor(q2, m, 32);
      }
      if (lane == 0) {
        float S = s + s2, Q = q + q2;
        float mean = S * (1.0f / HH);
        float var  = Q * (1.0f / HH) - mean * mean;
        st[((size_t)gg * BB + b) * 2 + 0] = mean;
        st[((size_t)gg * BB + b) * 2 + 1] = rsqrtf(var + EPSV);
      }
    }
    gbar(cnt, gen, NWG);

    // ---- phase E1: z, r, ah = xh + r*(hU_h); partial moments of ah ----
    if (wid < 128) {
      const int b = wid >> 2, seg = wid & 3;
      const float mz = st[(0 * BB + b) * 2 + 0], rz = st[(0 * BB + b) * 2 + 1];
      const float mr = st[(1 * BB + b) * 2 + 0], rr = st[(1 * BB + b) * 2 + 1];
      float s = 0.f, q = 0.f;
#pragma unroll
      for (int i = 0; i < 8; ++i) {
        int col = seg * 256 + i * 32 + lane;
        size_t xi = ((size_t)b * TT + t) * HH + col;
        float az = xz[xi] + acc[(0 * BB + b) * HH + col];
        float z  = sigmoidf((az - mz) * rz * g_z[col] + be_z[col]);
        float av = xr[xi] + acc[(1 * BB + b) * HH + col];
        float r  = sigmoidf((av - mr) * rr * g_r[col] + be_r[col]);
        float ah = xh[xi] + r * acc[(2 * BB + b) * HH + col];
        zb[(size_t)b * HH + col]  = z;
        ahb[(size_t)b * HH + col] = ah;
        s += ah; q += ah * ah;
      }
#pragma unroll
      for (int m = 1; m < 32; m <<= 1) { s += __shfl_xor(s, m, 32); q += __shfl_xor(q, m, 32); }
      if (lane == 0) { p2s[b * 4 + seg] = s; p2q[b * 4 + seg] = q; }
    }
    gbar(cnt, gen, NWG);

    // ---- phase S2: LN stats for ah rows ----
    if (wid < 32) {
      const int b = wid;
      float s = (lane < 4) ? p2s[b * 4 + lane] : 0.f;
      float q = (lane < 4) ? p2q[b * 4 + lane] : 0.f;
      s += __shfl_xor(s, 1, 32); q += __shfl_xor(q, 1, 32);
      s += __shfl_xor(s, 2, 32); q += __shfl_xor(q, 2, 32);
      if (lane == 0) {
        float mean = s * (1.0f / HH);
        float var  = q * (1.0f / HH) - mean * mean;
        st[((size_t)2 * BB + b) * 2 + 0] = mean;
        st[((size_t)2 * BB + b) * 2 + 1] = rsqrtf(var + EPSV);
      }
    }
    gbar(cnt, gen, NWG);

    // ---- phase E2: h_new = z*h + (1-z)*sigmoid(LN(ah)); write out & h ----
    if (wid < 128) {
      const int b = wid >> 2, seg = wid & 3;
      const float mh = st[(2 * BB + b) * 2 + 0], rh = st[(2 * BB + b) * 2 + 1];
#pragma unroll
      for (int i = 0; i < 8; ++i) {
        int col = seg * 256 + i * 32 + lane;
        float ah = ahb[(size_t)b * HH + col];
        float hh = sigmoidf((ah - mh) * rh * g_h[col] + be_h[col]);
        float z  = zb[(size_t)b * HH + col];
        float ho = hf[(size_t)b * HH + col];
        float hn = z * ho + (1.0f - z) * hh;
        out[((size_t)b * TT + t) * HH + col] = hn;
        hf[(size_t)b * HH + col]  = hn;
        hbf[(size_t)b * HH + col] = f2bf(hn);
        if (t == TT - 1) hlast[(size_t)b * HH + col] = hn;
      }
    }
    gbar(cnt, gen, NWG);
  }
}

// ---------------- host launcher ----------------
extern "C" void kernel_launch(void* const* d_in, const int* in_sizes, int n_in,
                              void* d_out, int out_size, void* d_ws, size_t ws_size,
                              hipStream_t stream) {
  const float* x   = (const float*)d_in[0];
  const float* Wz  = (const float*)d_in[1];
  const float* Wr  = (const float*)d_in[2];
  const float* Wh  = (const float*)d_in[3];
  const float* Uz  = (const float*)d_in[4];
  const float* Ur  = (const float*)d_in[5];
  const float* Uh  = (const float*)d_in[6];
  const float* bz  = (const float*)d_in[7];
  const float* br  = (const float*)d_in[8];
  const float* h0  = (const float*)d_in[9];
  const float* gz  = (const float*)d_in[10];
  const float* bez = (const float*)d_in[11];
  const float* gr  = (const float*)d_in[12];
  const float* ber = (const float*)d_in[13];
  const float* gh  = (const float*)d_in[14];
  const float* beh = (const float*)d_in[15];

  char* ws = (char*)d_ws;
  float* xz   = (float*)(ws + OFF_XZ);
  float* xr   = (float*)(ws + OFF_XR);
  float* xh   = (float*)(ws + OFF_XH);
  bf16*  xbf  = (bf16*)(ws + OFF_XBF);
  bf16*  wp   = (bf16*)(ws + OFF_WP);
  bf16*  up   = (bf16*)(ws + OFF_UP);
  float* hf   = (float*)(ws + OFF_HF);
  bf16*  hbf  = (bf16*)(ws + OFF_HBF);
  float* acc  = (float*)(ws + OFF_ACC);
  float* zb   = (float*)(ws + OFF_ZB);
  float* ahb  = (float*)(ws + OFF_AHB);
  float* pps  = (float*)(ws + OFF_PS);
  float* ppq  = (float*)(ws + OFF_PQ);
  float* p2s  = (float*)(ws + OFF_P2S);
  float* p2q  = (float*)(ws + OFF_P2Q);
  float* st   = (float*)(ws + OFF_ST);
  int*   bar  = (int*)(ws + OFF_BAR);

  float* out   = (float*)d_out;
  float* hlast = out + (size_t)BB * TT * HH;

  // prep
  k_cvt_x<<<(BT * DD + 255) / 256, 256, 0, stream>>>(x, xbf, BT * DD);
  k_pack_w<<<(3 * 64 * 16 * 512 + 255) / 256, 256, 0, stream>>>(Wz, Wr, Wh, wp);
  k_pack_u<<<(3 * 64 * 32 * 512 + 255) / 256, 256, 0, stream>>>(Uz, Ur, Uh, up);
  k_init<<<(BB * HH + 255) / 256, 256, 0, stream>>>(h0, hf, hbf, bar);

  // input projections: 3 gates * 1024 m-strips * 64 n-tiles waves / 8 waves per block
  k_proj<<<(3 * 1024 * 64) / 8, 256, 0, stream>>>(xbf, wp, bz, br, xz, xr, xh);

  // persistent recurrent scan
  k_scan<<<NWG, 128, 0, stream>>>(hbf, hf, up, xz, xr, xh, acc, zb, ahb,
                                  pps, ppq, p2s, p2q, st, bar,
                                  gz, bez, gr, ber, gh, beh, out, hlast);
}
